// ESN_13812614824406
// MI455X (gfx1250) — matrix operand order repaired and checked by reading, not compiled, not run on hardware
//
#include <hip/hip_runtime.h>

// ---------------- problem constants (match reference) ----------------
#define N_RES 2048
#define T_TOT 8192
#define WARM  512
#define TR    (T_TOT - WARM)      // 7680
#define O_OUT 2
#define LEAK  0.3f
#define REGL  1e-6f

#define NB_REC 32                 // blocks in recurrence (32 x 64 = 2048 rows)
#define BT_REC 64
#define CG_ITERS 512
#define JT 4                      // j-tiles per wave in the Gram kernel

typedef __attribute__((ext_vector_type(16))) __bf16          v16bf;
typedef __attribute__((ext_vector_type(8)))  unsigned short  u16x8;
typedef __attribute__((ext_vector_type(16))) unsigned short  u16x16;
typedef __attribute__((ext_vector_type(8)))  float           v8f;

// ---------------- bf16 helpers (bit-exact RNE, no __bf16 arithmetic) --------
__device__ __forceinline__ unsigned short f2bf(float f) {
    unsigned u = __float_as_uint(f);
    unsigned r = u + 0x7FFFu + ((u >> 16) & 1u);   // round-to-nearest-even
    return (unsigned short)(r >> 16);
}
__device__ __forceinline__ float bf2f(unsigned short h) {
    return __uint_as_float(((unsigned)h) << 16);
}

// ---------------- grid-wide barrier (monotonic counter) ---------------------
__device__ __forceinline__ void grid_sync(unsigned* cnt, unsigned target) {
    __threadfence();           // make this thread's global writes visible
    __syncthreads();
    if (threadIdx.x == 0) {
        atomicAdd(cnt, 1u);
        while (atomicAdd(cnt, 0u) < target) { __builtin_amdgcn_s_sleep(1); }
        __threadfence();
    }
    __syncthreads();
}

// ---------------- K0: W -> W^T in [k/4][n][k%4] panel layout ----------------
// Lets the recurrence matvec read one float4 (b128) per 4 k-values, coalesced
// across lanes (lane n -> consecutive 16B).
__global__ void esn_transpose(const float* __restrict__ W, float* __restrict__ Wt4) {
    int n = blockIdx.x;
    for (int k4 = threadIdx.x; k4 < N_RES / 4; k4 += blockDim.x) {
        float4 w = *(const float4*)(W + (size_t)n * N_RES + (size_t)k4 * 4);
        *(float4*)(Wt4 + (size_t)k4 * N_RES * 4 + (size_t)n * 4) = w;
    }
}

// ---------------- K1: drive[t][n] = Win u_t + Wf y_t + Wb -------------------
__global__ void esn_drive(const float* __restrict__ u, const float* __restrict__ y,
                          const float* __restrict__ Win, const float* __restrict__ Wf,
                          const float* __restrict__ Wb, float* __restrict__ driveX) {
    int t = blockIdx.x;
    float ut = u[t], y0 = y[t], y1 = y[T_TOT + t];
    for (int n = threadIdx.x; n < N_RES; n += blockDim.x) {
        float d = fmaf(Win[n], ut, fmaf(Wf[2 * n], y0, fmaf(Wf[2 * n + 1], y1, Wb[n])));
        driveX[(size_t)t * N_RES + n] = d;
    }
}

// ---------------- K2: sequential ESN recurrence (persistent grid) -----------
// driveX[t][:] is consumed at step t and overwritten in place with X[t][:].
__global__ __launch_bounds__(BT_REC) void esn_recurrence(
    const float* __restrict__ Wt4, float* __restrict__ driveX,
    float* __restrict__ xbuf, unsigned* __restrict__ cnt) {
    __shared__ __align__(16) float xs[N_RES];
    const int tid = threadIdx.x;
    const int n = blockIdx.x * BT_REC + tid;      // this thread's row
    for (int t = 0; t < T_TOT; ++t) {
        const int p = t & 1;
        const float* xin = xbuf + p * N_RES;
        for (int k = tid; k < N_RES; k += BT_REC) xs[k] = xin[k];
        __syncthreads();
        float a0 = 0.f, a1 = 0.f, a2 = 0.f, a3 = 0.f;
        #pragma unroll 2
        for (int k4 = 0; k4 < N_RES / 4; ++k4) {
            float4 w  = *(const float4*)(Wt4 + (size_t)k4 * N_RES * 4 + (size_t)n * 4);
            float4 xv = *(const float4*)(&xs[k4 * 4]);    // ds_load_b128 broadcast
            a0 = fmaf(w.x, xv.x, a0);
            a1 = fmaf(w.y, xv.y, a1);
            a2 = fmaf(w.z, xv.z, a2);
            a3 = fmaf(w.w, xv.w, a3);
        }
        float acc = (a0 + a1) + (a2 + a3);
        float d  = driveX[(size_t)t * N_RES + n];
        float xn = (1.0f - LEAK) * xs[n] + LEAK * tanhf(acc + d);
        xbuf[(1 - p) * N_RES + n] = xn;
        driveX[(size_t)t * N_RES + n] = xn;       // X[t][n]
        grid_sync(cnt, (unsigned)NB_REC * (unsigned)(t + 1));
    }
}

// ---------------- K3: fp32 X -> bf16 hi/lo in WMMA K-panel layout -----------
// Element (t, n) stored at  (t/32)*32*N + n*32 + (t%32)  (u16 units).
// This makes every WMMA operand two contiguous 16B runs per lane.
__global__ void esn_bf16split(const float* __restrict__ X,
                              unsigned short* __restrict__ Xh,
                              unsigned short* __restrict__ Xl) {
    int panel = blockIdx.x;                        // 0..TR/32-1
    for (int tt = 0; tt < 32; ++tt) {
        int t = panel * 32 + tt;
        for (int n = threadIdx.x; n < N_RES; n += blockDim.x) {
            float x = X[(size_t)(WARM + t) * N_RES + n];
            unsigned short h = f2bf(x);
            float lo = x - bf2f(h);
            size_t o = (size_t)panel * 32 * N_RES + (size_t)n * 32 + tt;
            Xh[o] = h;
            Xl[o] = f2bf(lo);
        }
    }
}

// Load one WMMA 16-bit operand: two contiguous 8-element (16B) runs per lane.
__device__ __forceinline__ v16bf ld_op(const unsigned short* __restrict__ base,
                                       size_t off_lo, size_t off_hi) {
    u16x8 lo = *(const u16x8*)(base + off_lo);
    u16x8 hi = *(const u16x8*)(base + off_hi);
    u16x16 w = __builtin_shufflevector(lo, hi, 0, 1, 2, 3, 4, 5, 6, 7,
                                               8, 9, 10, 11, 12, 13, 14, 15);
    return __builtin_bit_cast(v16bf, w);
}

// ---------------- K4: A = Xeff Xeff^T + reg I via bf16 WMMA -----------------
// One wave computes JT 16x16 C tiles along j (A-operand reuse).
// A ~= Xh Xh^T + Xh Xl^T + Xl Xh^T   (fp32 accumulate), 3*JT WMMAs / k-step.
__global__ __launch_bounds__(32) void esn_gram(
    const unsigned short* __restrict__ Xh, const unsigned short* __restrict__ Xl,
    float* __restrict__ Aout) {
    const int lane = threadIdx.x;
    const int half = lane >> 4;
    const int l4   = lane & 15;
    const int gi0 = blockIdx.y * 16;
    const int ia  = gi0 + l4;                       // A-operand row (M = l4)
    v8f c[JT] = {};
    for (int kb = 0; kb < TR; kb += 32) {
        const size_t pb = (size_t)kb * N_RES;       // panel base (u16 units)
        // A operand: K run of 16 starting at K = 16*half
        const size_t arow = pb + (size_t)ia * 32;
        v16bf ahv = ld_op(Xh, arow + 16 * half, arow + 16 * half + 8);
        v16bf alv = ld_op(Xl, arow + 16 * half, arow + 16 * half + 8);
        #pragma unroll
        for (int jj = 0; jj < JT; ++jj) {
            const int jb = (blockIdx.x * JT + jj) * 16 + l4;   // B col (N = l4)
            const size_t brow = pb + (size_t)jb * 32 + 16 * half; // K run of 16
            v16bf bhv = ld_op(Xh, brow, brow + 8);
            v16bf blv = ld_op(Xl, brow, brow + 8);
            c[jj] = __builtin_amdgcn_wmma_f32_16x16x32_bf16(false, ahv, false, bhv, (short)0, c[jj], false, false);
            c[jj] = __builtin_amdgcn_wmma_f32_16x16x32_bf16(false, ahv, false, blv, (short)0, c[jj], false, false);
            c[jj] = __builtin_amdgcn_wmma_f32_16x16x32_bf16(false, alv, false, bhv, (short)0, c[jj], false, false);
        }
    }
    #pragma unroll
    for (int jj = 0; jj < JT; ++jj) {
        const int gj = (blockIdx.x * JT + jj) * 16 + l4;
        #pragma unroll
        for (int r = 0; r < 8; ++r) {
            int gi = gi0 + r + 8 * half;
            float v = c[jj][r] + ((gi == gj) ? REGL : 0.0f);
            Aout[(size_t)gi * N_RES + gj] = v;
        }
    }
}

// ---------------- K5: B = Yt Xeff^T  (tiny, fp32) ---------------------------
__global__ void esn_bmat(const float* __restrict__ X, const float* __restrict__ Yt,
                         float* __restrict__ Bm) {
    int n = blockIdx.x * blockDim.x + threadIdx.x;
    if (n >= N_RES) return;
    float b0 = 0.f, b1 = 0.f;
    for (int t = 0; t < TR; ++t) {
        float x = X[(size_t)(WARM + t) * N_RES + n];
        b0 = fmaf(Yt[t], x, b0);
        b1 = fmaf(Yt[TR + t], x, b1);
    }
    Bm[n] = b0;
    Bm[N_RES + n] = b1;
}

// ---------------- K6: CG solve (A SPD) for both RHS, one workgroup ----------
__global__ __launch_bounds__(1024) void esn_cg(
    const float* __restrict__ A, const float* __restrict__ Bm,
    float* __restrict__ out) {
    __shared__ float pv[N_RES];
    __shared__ float red[1024];
    __shared__ float s_rs, s_pap, s_rsn;
    const int tid = threadIdx.x;
    const int n0 = tid, n1 = tid + 1024;
    for (int o = 0; o < O_OUT; ++o) {
        __syncthreads();
        float b0 = Bm[o * N_RES + n0], b1 = Bm[o * N_RES + n1];
        float x0 = 0.f, x1 = 0.f, r0 = b0, r1 = b1;
        pv[n0] = b0; pv[n1] = b1;
        red[tid] = b0 * b0 + b1 * b1;
        __syncthreads();
        for (int s = 512; s > 0; s >>= 1) { if (tid < s) red[tid] += red[tid + s]; __syncthreads(); }
        if (tid == 0) s_rs = red[0];
        __syncthreads();
        for (int it = 0; it < CG_ITERS; ++it) {
            // Ap: exploit symmetry -> column-major access is coalesced
            float ap0 = 0.f, ap1 = 0.f;
            for (int k = 0; k < N_RES; ++k) {
                float pk = pv[k];
                ap0 = fmaf(A[(size_t)k * N_RES + n0], pk, ap0);
                ap1 = fmaf(A[(size_t)k * N_RES + n1], pk, ap1);
            }
            red[tid] = pv[n0] * ap0 + pv[n1] * ap1;
            __syncthreads();
            for (int s = 512; s > 0; s >>= 1) { if (tid < s) red[tid] += red[tid + s]; __syncthreads(); }
            if (tid == 0) s_pap = red[0];
            __syncthreads();
            float den = (fabsf(s_pap) > 1e-30f) ? s_pap : 1e-30f;
            float alpha = s_rs / den;
            x0 = fmaf(alpha, pv[n0], x0);
            x1 = fmaf(alpha, pv[n1], x1);
            r0 = fmaf(-alpha, ap0, r0);
            r1 = fmaf(-alpha, ap1, r1);
            red[tid] = r0 * r0 + r1 * r1;
            __syncthreads();
            for (int s = 512; s > 0; s >>= 1) { if (tid < s) red[tid] += red[tid + s]; __syncthreads(); }
            if (tid == 0) s_rsn = red[0];
            __syncthreads();
            float rden = (fabsf(s_rs) > 1e-30f) ? s_rs : 1e-30f;
            float beta = s_rsn / rden;
            pv[n0] = fmaf(beta, pv[n0], r0);
            pv[n1] = fmaf(beta, pv[n1], r1);
            if (tid == 0) s_rs = s_rsn;
            __syncthreads();
        }
        out[o * N_RES + n0] = x0;
        out[o * N_RES + n1] = x1;
    }
}

// ---------------- host launcher ---------------------------------------------
extern "C" void kernel_launch(void* const* d_in, const int* in_sizes, int n_in,
                              void* d_out, int out_size, void* d_ws, size_t ws_size,
                              hipStream_t stream) {
    (void)in_sizes; (void)n_in; (void)out_size; (void)ws_size;
    const float* u   = (const float*)d_in[0];   // [1, 8192]
    const float* y   = (const float*)d_in[1];   // [2, 8192]
    const float* Yt  = (const float*)d_in[2];   // [2, 7680]
    const float* Win = (const float*)d_in[3];   // [2048, 1]
    const float* W   = (const float*)d_in[4];   // [2048, 2048]
    const float* Wf  = (const float*)d_in[5];   // [2048, 2]
    const float* Wb  = (const float*)d_in[6];   // [2048]
    float* out = (float*)d_out;                 // [2, 2048]

    // workspace layout (bytes)
    char* ws = (char*)d_ws;
    unsigned* cnt = (unsigned*)(ws);                                   // barrier counter
    float* xbuf   = (float*)(ws + 1024);                               // 2 x 2048 f32
    float* Wt4    = (float*)(ws + (size_t)(1 << 16));                  // 16 MB, panel layout
    float* driveX = (float*)((char*)Wt4 + (size_t)N_RES * N_RES * 4);  // 64 MB (drive, then X)
    unsigned short* Xh = (unsigned short*)((char*)driveX + (size_t)T_TOT * N_RES * 4); // 31.5 MB
    unsigned short* Xl = (unsigned short*)((char*)Xh + (size_t)TR * N_RES * 2);        // 31.5 MB
    float* Aout   = (float*)((char*)Xl + (size_t)TR * N_RES * 2);      // 16 MB
    float* Bm     = (float*)((char*)Aout + (size_t)N_RES * N_RES * 4); // 16 KB

    hipMemsetAsync(d_ws, 0, (size_t)(1 << 16), stream);  // zero counter + x state

    esn_transpose<<<N_RES, 256, 0, stream>>>(W, Wt4);
    esn_drive<<<T_TOT, 256, 0, stream>>>(u, y, Win, Wf, Wb, driveX);
    esn_recurrence<<<NB_REC, BT_REC, 0, stream>>>(Wt4, driveX, xbuf, cnt);
    esn_bf16split<<<TR / 32, 256, 0, stream>>>(driveX, Xh, Xl);
    dim3 gg(N_RES / 16 / JT, N_RES / 16);
    esn_gram<<<gg, 32, 0, stream>>>(Xh, Xl, Aout);
    esn_bmat<<<(N_RES + 255) / 256, 256, 0, stream>>>(driveX, Yt, Bm);
    esn_cg<<<1, 1024, 0, stream>>>(Aout, Bm, out);
}
